// GATLayer_43593918055149
// MI455X (gfx1250) — matrix operand order repaired and checked by reading, not compiled, not run on hardware
//
#include <hip/hip_runtime.h>
#include <hip/hip_bf16.h>
#include <stdint.h>

typedef __attribute__((ext_vector_type(16))) __bf16 v16bf;
typedef __attribute__((ext_vector_type(8)))  float  v8f;

#define NN   8192
#define FIN  128
#define NH   4
#define DD   32
#define FH   128   /* NH*DD */
#define NCH  (NN / 32)

// ---------------------------------------------------------------------------
// K1: Wh = h @ W  (f32, LDS-tiled). Writes WhT_bf16[c][n] (c = h*32+d) and
//     s_src[h][n], s_dst[h][n].  128 threads, 32 rows per block.
// ---------------------------------------------------------------------------
__global__ __launch_bounds__(128) void wh_kernel(
    const float* __restrict__ hmat, const float* __restrict__ Wmat,
    const float* __restrict__ avec, uint16_t* __restrict__ whT,
    float* __restrict__ ssg, float* __restrict__ sdg) {
  __shared__ float s_h[32][33];
  __shared__ float s_W[32][129];
  const int t  = threadIdx.x;
  const int i0 = blockIdx.x * 32;
  const int r  = t >> 2;          // 0..31 : row of this thread's outputs
  const int cg = t & 3;           // head / 32-col group
  const int c0 = cg * 32;

  float acc[32];
#pragma unroll
  for (int cc = 0; cc < 32; ++cc) acc[cc] = 0.f;

  for (int kb = 0; kb < FIN; kb += 32) {
    { // stage h tile: 32 rows x 32 k
      int row = t >> 2, ks = (t & 3) * 8;
      float4 h0 = *(const float4*)(hmat + (size_t)(i0 + row) * FIN + kb + ks);
      float4 h1 = *(const float4*)(hmat + (size_t)(i0 + row) * FIN + kb + ks + 4);
      s_h[row][ks + 0] = h0.x; s_h[row][ks + 1] = h0.y;
      s_h[row][ks + 2] = h0.z; s_h[row][ks + 3] = h0.w;
      s_h[row][ks + 4] = h1.x; s_h[row][ks + 5] = h1.y;
      s_h[row][ks + 6] = h1.z; s_h[row][ks + 7] = h1.w;
      int kk = t >> 2, cw = (t & 3) * 32;
#pragma unroll
      for (int q = 0; q < 8; ++q) {
        float4 wv = *(const float4*)(Wmat + (size_t)(kb + kk) * FH + cw + q * 4);
        s_W[kk][cw + q * 4 + 0] = wv.x; s_W[kk][cw + q * 4 + 1] = wv.y;
        s_W[kk][cw + q * 4 + 2] = wv.z; s_W[kk][cw + q * 4 + 3] = wv.w;
      }
    }
    __syncthreads();
    for (int kk = 0; kk < 32; ++kk) {
      float hv = s_h[r][kk];
#pragma unroll
      for (int cc = 0; cc < 32; ++cc)
        acc[cc] = fmaf(hv, s_W[kk][c0 + cc], acc[cc]);
    }
    __syncthreads();
  }

  const int i = i0 + r;
  float ss = 0.f, sd = 0.f;
#pragma unroll
  for (int cc = 0; cc < 32; ++cc) {
    whT[(size_t)(c0 + cc) * NN + i] =
        __builtin_bit_cast(uint16_t, (__bf16)acc[cc]);   // hw v_cvt if present
    ss += acc[cc] * avec[cg * 64 + cc];        // a_src
    sd += acc[cc] * avec[cg * 64 + 32 + cc];   // a_dst
  }
  ssg[cg * NN + i] = ss;
  sdg[cg * NN + i] = sd;
}

// ---------------------------------------------------------------------------
// K2: per-head global max of s_dst  (4 blocks x 256 threads)
// ---------------------------------------------------------------------------
__global__ __launch_bounds__(256) void maxred_kernel(
    const float* __restrict__ sdg, float* __restrict__ Mw) {
  __shared__ float red[256];
  const int h = blockIdx.x, t = threadIdx.x;
  float m = -3.4e38f;
  for (int j = t; j < NN; j += 256) m = fmaxf(m, sdg[h * NN + j]);
  red[t] = m;
  __syncthreads();
  for (int s = 128; s > 0; s >>= 1) {
    if (t < s) red[t] = fmaxf(red[t], red[t + s]);
    __syncthreads();
  }
  if (t == 0) Mw[h] = red[0];
}

// ---------------------------------------------------------------------------
// K3: separable exp tables.  m_i = leakyrelu(ss_i + M_h) >= max_j e[i,j].
//   P(s>=0) = A1[i]*B1[j] = e^{ss+sd-m};  P(s<0) = A2[i]*B2[j] = e^{0.2(ss+sd)-m}
// ---------------------------------------------------------------------------
__global__ __launch_bounds__(256) void scale_kernel(
    const float* __restrict__ ssg, const float* __restrict__ sdg,
    const float* __restrict__ Mw, float* __restrict__ A1g,
    float* __restrict__ A2g, float* __restrict__ B1g, float* __restrict__ B2g) {
  const int n = blockIdx.x * 256 + threadIdx.x;  // 0..NH*NN-1
  const int h = n >> 13;
  const float ss = ssg[n], sd = sdg[n], M = Mw[h];
  const float z = ss + M;
  const float m = (z >= 0.f) ? z : 0.2f * z;
  A1g[n] = __expf(ss - m);
  A2g[n] = __expf(0.2f * ss - m);
  B1g[n] = __expf(sd);
  B2g[n] = __expf(0.2f * sd);
}

// ---------------------------------------------------------------------------
// Async staging of one 32-wide j chunk into LDS (CDNA5 ASYNCcnt path).
// All destinations / sources are 16B aligned for the b128 form.
// ---------------------------------------------------------------------------
__device__ __forceinline__ void stage_async(
    int tid, int i0, int j0,
    const int* __restrict__ adj, const uint16_t* __restrict__ whT,
    const float* __restrict__ sdg, const float* __restrict__ B1g,
    const float* __restrict__ B2g,
    int* sadj, uint32_t* swh, float* ssd, float* sb1, float* sb2) {
  // adj tile: 32 rows x 32 cols, 4 ints per thread
  {
    const int ar = tid >> 3, ac = (tid & 7) * 4;
    unsigned d = (unsigned)(uintptr_t)(sadj + ar * 36 + ac);
    unsigned long long g =
        (unsigned long long)(uintptr_t)(adj + (size_t)(i0 + ar) * NN + j0 + ac);
    asm volatile("global_load_async_to_lds_b128 %0, %1, off"
                 :: "v"(d), "v"(g) : "memory");
  }
  // Wh slice: 128 c-rows x 32 j (bf16), 2 threads per c-row, 2x16B each
  {
    const int c = tid >> 1, jg = (tid & 1) * 8;  // dword (pair) index
    unsigned d = (unsigned)(uintptr_t)(swh + c * 20 + jg);
    unsigned long long g =
        (unsigned long long)(uintptr_t)(whT + (size_t)c * NN + j0 + jg * 2);
    asm volatile("global_load_async_to_lds_b128 %0, %1, off"
                 :: "v"(d), "v"(g) : "memory");
    asm volatile("global_load_async_to_lds_b128 %0, %1, off"
                 :: "v"(d + 16u), "v"(g + 16ull) : "memory");
  }
  // per-j tables: 4 heads x 32, one f32 per thread per table
  if (tid < 128) {
    const int hh = tid >> 5, k = tid & 31;
    unsigned d0 = (unsigned)(uintptr_t)(ssd + hh * 32 + k);
    unsigned d1 = (unsigned)(uintptr_t)(sb1 + hh * 32 + k);
    unsigned d2 = (unsigned)(uintptr_t)(sb2 + hh * 32 + k);
    unsigned long long g0 =
        (unsigned long long)(uintptr_t)(sdg + (size_t)hh * NN + j0 + k);
    unsigned long long g1 =
        (unsigned long long)(uintptr_t)(B1g + (size_t)hh * NN + j0 + k);
    unsigned long long g2 =
        (unsigned long long)(uintptr_t)(B2g + (size_t)hh * NN + j0 + k);
    asm volatile("global_load_async_to_lds_b32 %0, %1, off" :: "v"(d0), "v"(g0) : "memory");
    asm volatile("global_load_async_to_lds_b32 %0, %1, off" :: "v"(d1), "v"(g1) : "memory");
    asm volatile("global_load_async_to_lds_b32 %0, %1, off" :: "v"(d2), "v"(g2) : "memory");
  }
}

__device__ __forceinline__ float pval(int av, float sd, float b1, float b2,
                                      float a1, float a2, float ssr) {
  const float c1 = a1 * b1;
  const float c2 = a2 * b2;
  float pv = (ssr + sd >= 0.f) ? c1 : c2;   // v_cndmask
  return (av != 0) ? pv : 0.f;              // v_cndmask
}

// ---------------------------------------------------------------------------
// K4: fused masked-softmax attention + aggregation.
//   256 blocks x 256 threads; 8 waves = 4 heads x 2 row-tiles of 16.
//   Double-buffered async LDS staging; 2x v_wmma_f32_16x16x32_bf16 per chunk.
// ---------------------------------------------------------------------------
__global__ __launch_bounds__(256) void gat_attn_kernel(
    const int* __restrict__ adj, const uint16_t* __restrict__ whT,
    const float* __restrict__ ssg, const float* __restrict__ A1g,
    const float* __restrict__ A2g, const float* __restrict__ B1g,
    const float* __restrict__ B2g, const float* __restrict__ sdg,
    float* __restrict__ out) {
  __shared__ int      s_adj[2][32][36];      // 16B-aligned runs
  __shared__ uint32_t s_wh32[2][128][20];    // bf16 pairs [c][j_pair]
  __shared__ float    s_sd[2][NH][32], s_B1[2][NH][32], s_B2[2][NH][32];
  __shared__ float    s_l[8][16];

  const int tid  = threadIdx.x;
  const int lane = tid & 31;
  const int w    = tid >> 5;      // 0..7
  const int h    = w & 3;         // head
  const int rt   = w >> 2;        // row-tile 0/1
  const int i0   = blockIdx.x * 32;
  const int r    = lane & 15;     // matrix row (A) / column (B,C)
  const bool hi  = lane >= 16;
  const int irow = i0 + rt * 16 + r;

  const float a1  = A1g[h * NN + irow];
  const float a2  = A2g[h * NN + irow];
  const float ssr = ssg[h * NN + irow];

  v8f acc0 = {0.f, 0.f, 0.f, 0.f, 0.f, 0.f, 0.f, 0.f};
  v8f acc1 = {0.f, 0.f, 0.f, 0.f, 0.f, 0.f, 0.f, 0.f};
  float lsum = 0.f;

  const int khalf = hi ? 8 : 0;
  const int arow  = rt * 16 + r;
  const int kofs[4] = {0, 4, 16, 20};   // A-layout K runs (pairs u[2g],u[2g+1])

  // prologue: issue chunk 0 into buffer 0
  stage_async(tid, i0, 0, adj, whT, sdg, B1g, B2g,
              &s_adj[0][0][0], &s_wh32[0][0][0],
              &s_sd[0][0][0], &s_B1[0][0][0], &s_B2[0][0][0]);

  for (int c = 0; c < NCH; ++c) {
    const int cur = c & 1;
    asm volatile("s_wait_asynccnt 0x0" ::: "memory");
    __syncthreads();                       // all buffers for chunk c visible

    if (c + 1 < NCH) {                     // issue next chunk into other buffer
      const int nb = (c + 1) & 1;
      stage_async(tid, i0, (c + 1) * 32, adj, whT, sdg, B1g, B2g,
                  &s_adj[nb][0][0], &s_wh32[nb][0][0],
                  &s_sd[nb][0][0], &s_B1[nb][0][0], &s_B2[nb][0][0]);
    }

    // ---- wide LDS loads for this lane's A-layout K runs ----
    int4   adjv[4];
    float4 sdv[4], b1v[4], b2v[4];
#pragma unroll
    for (int g = 0; g < 4; ++g) {
      const int kb = khalf + kofs[g];
      adjv[g] = *(const int4*)  &s_adj[cur][arow][kb];
      sdv[g]  = *(const float4*)&s_sd[cur][h][kb];
      b1v[g]  = *(const float4*)&s_B1[cur][h][kb];
      b2v[g]  = *(const float4*)&s_B2[cur][h][kb];
    }

    // ---- build A fragment (P, 16x32 bf16), branch-free selects ----
    v16bf Av;
#pragma unroll
    for (int g = 0; g < 4; ++g) {
      const float p0 = pval(adjv[g].x, sdv[g].x, b1v[g].x, b2v[g].x, a1, a2, ssr);
      const float p1 = pval(adjv[g].y, sdv[g].y, b1v[g].y, b2v[g].y, a1, a2, ssr);
      const float p2 = pval(adjv[g].z, sdv[g].z, b1v[g].z, b2v[g].z, a1, a2, ssr);
      const float p3 = pval(adjv[g].w, sdv[g].w, b1v[g].w, b2v[g].w, a1, a2, ssr);
      lsum += (p0 + p1) + (p2 + p3);
      Av[4 * g + 0] = (__bf16)p0;
      Av[4 * g + 1] = (__bf16)p1;
      Av[4 * g + 2] = (__bf16)p2;
      Av[4 * g + 3] = (__bf16)p3;
    }

    // ---- B fragments (Wh chunk, 32x16 bf16): two b128 LDS loads each ----
    const int pbase = hi ? 4 : 0;
    union { int4 q[2]; v16bf v; } B0f, B1f;
    B0f.q[0] = *(const int4*)&s_wh32[cur][h * 32 + r][pbase];
    B0f.q[1] = *(const int4*)&s_wh32[cur][h * 32 + r][pbase + 8];
    B1f.q[0] = *(const int4*)&s_wh32[cur][h * 32 + 16 + r][pbase];
    B1f.q[1] = *(const int4*)&s_wh32[cur][h * 32 + 16 + r][pbase + 8];

    acc0 = __builtin_amdgcn_wmma_f32_16x16x32_bf16(
        false, Av, false, B0f.v, (short)0, acc0, false, false);
    acc1 = __builtin_amdgcn_wmma_f32_16x16x32_bf16(
        false, Av, false, B1f.v, (short)0, acc1, false, false);
  }

  // ---- row sums: lanes r and r+16 hold disjoint K halves
  lsum += __shfl_xor(lsum, 16, 32);
  if (!hi) s_l[w][r] = lsum;
  __syncthreads();

  // ---- epilogue: normalize, ELU, store (C layout: reg t -> row t + 8*hi)
#pragma unroll
  for (int t = 0; t < 8; ++t) {
    const int row = t + (hi ? 8 : 0);
    const int i = i0 + rt * 16 + row;
    float lr = s_l[w][row];
    lr = (lr > 0.f) ? lr : 1.f;
    float x0 = acc0[t] / lr;
    float x1 = acc1[t] / lr;
    x0 = (x0 > 0.f) ? x0 : (__expf(x0) - 1.f);
    x1 = (x1 > 0.f) ? x1 : (__expf(x1) - 1.f);
    out[(size_t)i * FH + h * 32 + r]      = x0;
    out[(size_t)i * FH + h * 32 + 16 + r] = x1;
  }
}

// ---------------------------------------------------------------------------
extern "C" void kernel_launch(void* const* d_in, const int* in_sizes, int n_in,
                              void* d_out, int out_size, void* d_ws, size_t ws_size,
                              hipStream_t stream) {
  const float* h   = (const float*)d_in[0];
  const int*   adj = (const int*)d_in[1];
  const float* W   = (const float*)d_in[2];
  const float* a   = (const float*)d_in[3];
  float* out = (float*)d_out;

  // workspace layout (~2.9 MB)
  uint8_t* ws = (uint8_t*)d_ws;
  uint16_t* whT = (uint16_t*)(ws);                       // bf16 [128][8192] = 2 MB
  float* ssg = (float*)(ws + 2097152);                   // [4][8192] each below
  float* sdg = (float*)(ws + 2097152 + 1 * 131072);
  float* A1g = (float*)(ws + 2097152 + 2 * 131072);
  float* A2g = (float*)(ws + 2097152 + 3 * 131072);
  float* B1g = (float*)(ws + 2097152 + 4 * 131072);
  float* B2g = (float*)(ws + 2097152 + 5 * 131072);
  float* Mw  = (float*)(ws + 2097152 + 6 * 131072);

  wh_kernel<<<NN / 32, 128, 0, stream>>>(h, W, a, whT, ssg, sdg);
  maxred_kernel<<<NH, 256, 0, stream>>>(sdg, Mw);
  scale_kernel<<<(NH * NN) / 256, 256, 0, stream>>>(ssg, sdg, Mw, A1g, A2g, B1g, B2g);
  gat_attn_kernel<<<NN / 32, 256, 0, stream>>>(adj, whT, ssg, A1g, A2g, B1g, B2g,
                                               sdg, out);
}
